// Decoder_79809082294812
// MI455X (gfx1250) — matrix operand order repaired and checked by reading, not compiled, not run on hardware
//
#include <hip/hip_runtime.h>
#include <hip/hip_bf16.h>

#define B_ 1024
#define N_ 1000
#define E_ 128
#define NPAD 1008   // N rounded up to 16-tile multiple

typedef __attribute__((ext_vector_type(16))) __bf16 bf16x16;
typedef __attribute__((ext_vector_type(8)))  float  f32x8;

__device__ __forceinline__ f32x8 wmma_bf16(bf16x16 a, bf16x16 b, f32x8 c) {
  // (neg_a, A, neg_b, B, c_mod, C, reuse_a, reuse_b)
  return __builtin_amdgcn_wmma_f32_16x16x32_bf16(false, a, false, b, (short)0, c, false, false);
}

// A-matrix (16x32 bf16) fragment per ISA table:
// lane<16 : M=lane,    elems 0..7 -> K=kc*32+0..7,  elems 8..15 -> K=kc*32+16..23
// lane>=16: M=lane-16, elems 0..7 -> K=kc*32+8..15, elems 8..15 -> K=kc*32+24..31
__device__ __forceinline__ bf16x16 load_a_frag(const float* __restrict__ row, int kc, int lane) {
  bf16x16 a;
  if (row) {
    int kb = kc * 32 + ((lane >> 4) << 3);
#pragma unroll
    for (int i = 0; i < 8; ++i) a[i] = (__bf16)row[kb + i];
#pragma unroll
    for (int i = 0; i < 8; ++i) a[i + 8] = (__bf16)row[kb + 16 + i];
  } else {
#pragma unroll
    for (int i = 0; i < 16; ++i) a[i] = (__bf16)0.0f;
  }
  return a;
}

// One workgroup (8 wave32s) per batch element. All passes over this batch's
// 512 KB embedding slice happen back-to-back -> passes 2..4 hit L2.
__global__ __launch_bounds__(256) void k_decoder(
    const float* __restrict__ emb, const float* __restrict__ cap,
    const float* __restrict__ Wqg, const float* __restrict__ Wkg,
    const float* __restrict__ Wvg, const float* __restrict__ Wog,
    const float* __restrict__ Wqo, const float* __restrict__ Wko,
    const int* __restrict__ cur, const unsigned char* __restrict__ mask,
    float* __restrict__ probs, float* __restrict__ logits) {
  int b = blockIdx.x, t = threadIdx.x;
  int wave = t >> 5, lane = t & 31;

  __shared__ float compat[NPAD * 9];   // compat/attn (stride 9); reused as lg[] in S4
  __shared__ float wembS[2][8][128];
  __shared__ float ctx[257];
  __shared__ float qv[128];
  __shared__ float qkS[128][8];        // qk_eff, with 1/sqrt(D) folded
  __shared__ float headsS[128], glimS[128], qoS[128], weffS[128];
  __shared__ float red[128];
  __shared__ float red8[8];

  const float* eb = emb + (size_t)b * N_ * E_;
  const unsigned char* mk = mask + (size_t)b * N_;

  // ---- S0: graph mean, context, q = ctx@Wqg, qk_eff[c][h] ----------------
  {
    int c = t & 127, half = t >> 7;
    float acc = 0.f;
    for (int n = half; n < N_; n += 2) acc += eb[(size_t)n * E_ + c];
    if (half) red[c] = acc;
    __syncthreads();
    if (!half) ctx[c] = (acc + red[c]) * (1.0f / (float)N_);
    int cn = cur[b];
    if (half) ctx[128 + c] = eb[(size_t)cn * E_ + c];
    if (t == 0) ctx[256] = cap[b];
    __syncthreads();

    if (t < 128) {                     // q[e] = sum_j ctx[j] * Wqg[j,e]
      float s = 0.f;
      for (int j = 0; j < 257; ++j) s = fmaf(ctx[j], Wqg[j * 128 + t], s);
      qv[t] = s;
    }
    __syncthreads();

    int h = t & 7;                     // qk_eff[c][h] = 0.25*sum_d Wkg[c,h*16+d]*q[h*16+d]
    for (int c0 = t >> 3; c0 < 128; c0 += 32) {
      float s = 0.f;
#pragma unroll
      for (int d = 0; d < 16; ++d) s = fmaf(Wkg[c0 * 128 + h * 16 + d], qv[h * 16 + d], s);
      qkS[c0][h] = s * 0.25f;
    }
    __syncthreads();
  }

  // ---- S1: compat[n][h] = emb[n,:] . qk_eff[:,h] via WMMA ----------------
  int col = lane & 15;
  int krow = (lane >> 4) << 3;         // B frag: lane<16 -> K 0..7,16..23, lane>=16 -> K 8..15,24..31
  int lm = lane & 15;
  {
    bf16x16 bfr[4];
#pragma unroll
    for (int kc = 0; kc < 4; ++kc)
#pragma unroll
      for (int i = 0; i < 16; ++i) {
        float v = (col < 8) ? qkS[kc * 32 + krow + i + ((i >> 3) << 3)][col] : 0.f;
        bfr[kc][i] = (__bf16)v;
      }

    for (int tile = wave; tile * 16 < N_; tile += 8) {
      int n0 = tile * 16;
      int nr = n0 + lm;
      const float* rowp = (nr < N_) ? (eb + (size_t)nr * E_) : nullptr;
      int nn = nr + 128;
      if (nn < N_) __builtin_prefetch(eb + (size_t)nn * E_, 0, 1);
      f32x8 acc = {0.f, 0.f, 0.f, 0.f, 0.f, 0.f, 0.f, 0.f};
#pragma unroll
      for (int kc = 0; kc < 4; ++kc) acc = wmma_bf16(load_a_frag(rowp, kc, lane), bfr[kc], acc);
      int h = lane & 15, mb = (lane >> 4) << 3;
      if (h < 8) {
#pragma unroll
        for (int j = 0; j < 8; ++j) {
          int n = n0 + mb + j;
          if (n < N_) compat[n * 9 + h] = mk[n] ? -3.0e38f : acc[j];
        }
      }
    }
    __syncthreads();
  }

  // ---- S2: softmax over n per head (wave w owns head w) ------------------
  {
    int h = wave;
    float mx = -3.4e38f;
    for (int n = lane; n < N_; n += 32) mx = fmaxf(mx, compat[n * 9 + h]);
#pragma unroll
    for (int o = 16; o; o >>= 1) mx = fmaxf(mx, __shfl_xor(mx, o, 32));
    float s = 0.f;
    for (int n = lane; n < N_; n += 32) s += __expf(compat[n * 9 + h] - mx);
#pragma unroll
    for (int o = 16; o; o >>= 1) s += __shfl_xor(s, o, 32);
    float inv = 1.0f / s;
    for (int n = lane; n < N_; n += 32)
      compat[n * 9 + h] = __expf(compat[n * 9 + h] - mx) * inv;
  }
  __syncthreads();

  // ---- S3: wemb[h][c] = sum_n attn[h][n]*emb[n][c]; chained matvecs ------
  {
    int c = t & 127, half = t >> 7;
    float a0 = 0, a1 = 0, a2 = 0, a3 = 0, a4 = 0, a5 = 0, a6 = 0, a7 = 0;
    for (int n = half; n < N_; n += 2) {
      float e = eb[(size_t)n * E_ + c];
      const float* at = &compat[n * 9];
      a0 = fmaf(at[0], e, a0); a1 = fmaf(at[1], e, a1);
      a2 = fmaf(at[2], e, a2); a3 = fmaf(at[3], e, a3);
      a4 = fmaf(at[4], e, a4); a5 = fmaf(at[5], e, a5);
      a6 = fmaf(at[6], e, a6); a7 = fmaf(at[7], e, a7);
    }
    wembS[half][0][c] = a0; wembS[half][1][c] = a1;
    wembS[half][2][c] = a2; wembS[half][3][c] = a3;
    wembS[half][4][c] = a4; wembS[half][5][c] = a5;
    wembS[half][6][c] = a6; wembS[half][7][c] = a7;
  }
  __syncthreads();

  if (t < 128) {   // heads[e'=t] = sum_c wemb[h][c] * Wvg[c, t]   (h = t>>4)
    int h = t >> 4;
    float s = 0.f;
    for (int c = 0; c < 128; ++c)
      s = fmaf(wembS[0][h][c] + wembS[1][h][c], Wvg[c * 128 + t], s);
    headsS[t] = s;
  }
  __syncthreads();
  if (t < 128) {   // glimpse = heads @ Wog
    float s = 0.f;
    for (int e = 0; e < 128; ++e) s = fmaf(headsS[e], Wog[e * 128 + t], s);
    glimS[t] = s;
  }
  __syncthreads();
  if (t < 128) {   // qo = glimpse @ Wqo
    float s = 0.f;
    for (int e = 0; e < 128; ++e) s = fmaf(glimS[e], Wqo[e * 128 + t], s);
    qoS[t] = s;
  }
  __syncthreads();
  if (t < 128) {   // w_eff[c] = (1/sqrt(E)) * sum_e Wko[c,e] * qo[e]
    float s = 0.f;
    for (int e = 0; e < 128; ++e) s = fmaf(Wko[t * 128 + e], qoS[e], s);
    weffS[t] = s * 0.08838834764831845f;
  }
  __syncthreads();

  // ---- S4: logits via WMMA (B col 0 = w_eff), tanh clip, row softmax -----
  float* lg = compat;                  // compat is dead; reuse first 1000 floats
  {
    bf16x16 bfr[4];
#pragma unroll
    for (int kc = 0; kc < 4; ++kc)
#pragma unroll
      for (int i = 0; i < 16; ++i)
        bfr[kc][i] = (__bf16)((col == 0) ? weffS[kc * 32 + krow + i + ((i >> 3) << 3)] : 0.f);

    for (int tile = wave; tile * 16 < N_; tile += 8) {
      int n0 = tile * 16;
      int nr = n0 + lm;
      const float* rowp = (nr < N_) ? (eb + (size_t)nr * E_) : nullptr;
      int nn = nr + 128;
      if (nn < N_) __builtin_prefetch(eb + (size_t)nn * E_, 0, 1);
      f32x8 acc = {0.f, 0.f, 0.f, 0.f, 0.f, 0.f, 0.f, 0.f};
#pragma unroll
      for (int kc = 0; kc < 4; ++kc) acc = wmma_bf16(load_a_frag(rowp, kc, lane), bfr[kc], acc);
      if ((lane & 15) == 0) {          // lanes 0 & 16 hold column 0 of D
        int mb = (lane >> 4) << 3;
#pragma unroll
        for (int j = 0; j < 8; ++j) {
          int n = n0 + mb + j;
          if (n < N_) {
            float l = 10.0f * tanhf(acc[j]);
            if (mk[n]) l = -__builtin_inff();
            lg[n] = l;
          }
        }
      }
    }
    __syncthreads();
  }

  float mx = -3.4e38f;
  for (int n = t; n < N_; n += 256) mx = fmaxf(mx, lg[n]);
#pragma unroll
  for (int o = 16; o; o >>= 1) mx = fmaxf(mx, __shfl_xor(mx, o, 32));
  if (lane == 0) red8[wave] = mx;
  __syncthreads();
  mx = red8[0];
#pragma unroll
  for (int w = 1; w < 8; ++w) mx = fmaxf(mx, red8[w]);
  __syncthreads();

  float s = 0.f;
  for (int n = t; n < N_; n += 256) s += __expf(lg[n] - mx);
#pragma unroll
  for (int o = 16; o; o >>= 1) s += __shfl_xor(s, o, 32);
  if (lane == 0) red8[wave] = s;
  __syncthreads();
  s = 0.f;
#pragma unroll
  for (int w = 0; w < 8; ++w) s += red8[w];
  float inv = 1.0f / s;

  for (int n = t; n < N_; n += 256) {
    float l = lg[n];
    probs[(size_t)b * N_ + n] = __expf(l - mx) * inv;
    logits[(size_t)b * N_ + n] = l;
  }
}

extern "C" void kernel_launch(void* const* d_in, const int* in_sizes, int n_in,
                              void* d_out, int out_size, void* d_ws, size_t ws_size,
                              hipStream_t stream) {
  const float* emb = (const float*)d_in[0];
  const float* cap = (const float*)d_in[1];
  const float* Wqg = (const float*)d_in[2];
  const float* Wkg = (const float*)d_in[3];
  const float* Wvg = (const float*)d_in[4];
  const float* Wog = (const float*)d_in[5];
  const float* Wqo = (const float*)d_in[6];
  const float* Wko = (const float*)d_in[7];
  const int* cur = (const int*)d_in[8];
  const unsigned char* mk = (const unsigned char*)d_in[9];

  float* probs = (float*)d_out;
  float* logits = probs + (size_t)B_ * N_;

  (void)d_ws; (void)ws_size; (void)in_sizes; (void)n_in; (void)out_size;

  k_decoder<<<B_, 256, 0, stream>>>(emb, cap, Wqg, Wkg, Wvg, Wog, Wqo, Wko,
                                    cur, mk, probs, logits);
}
